// CompiledConvLogicNet_6932077216081
// MI455X (gfx1250) — compile-verified, matrix operand order checked by reading, not run on hardware
//
#include <hip/hip_runtime.h>
#include <hip/hip_bf16.h>
#include <stdint.h>

// ---------------------------------------------------------------------------
// Binary logic-gate network, bit-packed over the batch dimension.
//  B=128 batch bits -> 4 x uint32 words per wire. Gates become bitwise ops
//  (op index 0..15 is the truth table: bit0<->f(1,1), bit1<->f(1,0),
//   bit2<->f(0,1), bit3<->f(0,0); verified against the COEF table).
//  Final GroupSum is done with V_WMMA_I32_16X16X64_IU8 against an all-ones
//  B matrix (every D column = row-sum), accumulated over 50 K-chunks.
// ---------------------------------------------------------------------------

typedef __attribute__((ext_vector_type(8))) int v8i;

#define NPIX   3072      // C*H*W
#define KK     64
#define PP     784
#define NKP    (KK*PP)   // 50176
#define NL1    64000
#define NL2    32000
#define NCLS   10
#define GRP    (NL2/NCLS) // 3200

__device__ __forceinline__ uint32_t gate32(uint32_t a, uint32_t b, int op) {
  uint32_t na = ~a, nb = ~b;
  uint32_t m0 = 0u - (uint32_t)( op       & 1);  // f(1,1)
  uint32_t m1 = 0u - (uint32_t)((op >> 1) & 1);  // f(1,0)
  uint32_t m2 = 0u - (uint32_t)((op >> 2) & 1);  // f(0,1)
  uint32_t m3 = 0u - (uint32_t)((op >> 3) & 1);  // f(0,0)
  return (m0 & a & b) | (m1 & a & nb) | (m2 & na & b) | (m3 & na & nb);
}

// ---- 1) bit-pack x[B=128, NPIX] floats -> xpk[NPIX][4] uint32 -------------
__global__ void k_bitpack(const float* __restrict__ x, uint32_t* __restrict__ xpk) {
  int pix = blockIdx.x * blockDim.x + threadIdx.x;  // 0..3071
  int w   = blockIdx.y;                             // 0..3
  if (pix >= NPIX) return;
  uint32_t word = 0;
  #pragma unroll 4
  for (int t = 0; t < 32; ++t) {
    float v = x[(size_t)(w * 32 + t) * NPIX + pix];
    word |= (v != 0.0f ? 1u : 0u) << t;
  }
  xpk[pix * 4 + w] = word;
}

// ---- 2) conv logic layer + 3-level gate tree ------------------------------
// thread = (kp = k*P+p, word w). Packed x staged in LDS (48KB).
__global__ void k_conv(const uint32_t* __restrict__ xpk,
                       const int* __restrict__ cl, const int* __restrict__ cr,
                       const int* __restrict__ o0, const int* __restrict__ o1,
                       const int* __restrict__ o2, const int* __restrict__ o3,
                       uint32_t* __restrict__ zconv) {
  __shared__ uint32_t sx[NPIX * 4];
  for (int i = threadIdx.x; i < NPIX * 4; i += blockDim.x) sx[i] = xpk[i];
  __syncthreads();

  int tid = threadIdx.x;
  int w   = tid & 3;
  int kp  = blockIdx.x * 64 + (tid >> 2);   // 784 blocks * 64 = 50176
  int k   = kp / PP;

  uint32_t v[8];
  #pragma unroll
  for (int g = 0; g < 8; ++g) {
    int ia = cl[kp * 8 + g];
    int ib = cr[kp * 8 + g];
    v[g] = gate32(sx[ia * 4 + w], sx[ib * 4 + w], o0[k * 8 + g]);
  }
  uint32_t t[4];
  #pragma unroll
  for (int j = 0; j < 4; ++j) t[j] = gate32(v[2*j], v[2*j+1], o1[k * 4 + j]);
  uint32_t u0 = gate32(t[0], t[1], o2[k * 2 + 0]);
  uint32_t u1 = gate32(t[2], t[3], o2[k * 2 + 1]);
  zconv[kp * 4 + w] = gate32(u0, u1, o3[k]);
}

// ---- 3) logic linear layer (packed -> packed) -----------------------------
__global__ void k_lin(const uint32_t* __restrict__ zin,
                      const int* __restrict__ ai, const int* __restrict__ bi,
                      const int* __restrict__ ops,
                      uint32_t* __restrict__ zout, int n) {
  int i = blockIdx.x * blockDim.x + threadIdx.x;
  int w = blockIdx.y;
  if (i >= n) return;
  zout[i * 4 + w] = gate32(zin[ai[i] * 4 + w], zin[bi[i] * 4 + w], ops[i]);
}

// ---- 4) logic linear layer 2, emitting bytes [B=128][NL2] row-major -------
__global__ void k_lin2_bytes(const uint32_t* __restrict__ z1,
                             const int* __restrict__ a2, const int* __restrict__ b2,
                             const int* __restrict__ p2,
                             uint8_t* __restrict__ z2b) {
  int i = blockIdx.x * blockDim.x + threadIdx.x;  // gate index
  int w = blockIdx.y;                             // batch word
  if (i >= NL2) return;
  uint32_t r = gate32(z1[a2[i] * 4 + w], z1[b2[i] * 4 + w], p2[i]);
  #pragma unroll
  for (int bit = 0; bit < 32; ++bit)
    z2b[(size_t)(w * 32 + bit) * NL2 + i] = (uint8_t)((r >> bit) & 1u);
}

// ---- 5) GroupSum via V_WMMA_I32_16X16X64_IU8 ------------------------------
// One full wave32 per (class c, batch tile bt). A = 16x64 byte tile of z2,
// B = all-ones bytes -> every D column equals the row sum. Accumulate 50
// chunks (3200 = 50*64), then lanes 0 and 16 hold column 0 (M=0..7 / 8..15).
__global__ void k_groupsum_wmma(const uint8_t* __restrict__ z2b,
                                float* __restrict__ out) {
  int c    = blockIdx.x / 8;   // class 0..9
  int bt   = blockIdx.x % 8;   // batch tile 0..7
  int lane = threadIdx.x;      // 0..31, full wave (EXEC all ones for WMMA)
  int m    = lane & 15;        // matrix row within tile
  int half = lane >> 4;        // K-half selector per ISA A-layout

  const uint8_t* rp = z2b + (size_t)(bt * 16 + m) * NL2 + (size_t)c * GRP;

  v8i ones, acc;
  #pragma unroll
  for (int i = 0; i < 8; ++i) { ones[i] = 0x01010101; acc[i] = 0; }

  for (int ch = 0; ch < GRP / 64; ++ch) {
    int base = ch * 64 + half * 8;                 // 8-byte aligned always
    uint2 q0 = *(const uint2*)(rp + base +  0);    // K {0..7}/{8..15}
    uint2 q1 = *(const uint2*)(rp + base + 16);    // K {16..23}/{24..31}
    uint2 q2 = *(const uint2*)(rp + base + 32);    // K {32..39}/{40..47}
    uint2 q3 = *(const uint2*)(rp + base + 48);    // K {48..55}/{56..63}
    v8i a;
    a[0] = (int)q0.x; a[1] = (int)q0.y;
    a[2] = (int)q1.x; a[3] = (int)q1.y;
    a[4] = (int)q2.x; a[5] = (int)q2.y;
    a[6] = (int)q3.x; a[7] = (int)q3.y;
    acc = __builtin_amdgcn_wmma_i32_16x16x64_iu8(
        /*sgn_a=*/false, a, /*sgn_b=*/false, ones, acc,
        /*reuse_a=*/false, /*reuse_b=*/false);
  }

  if ((lane & 15) == 0) {                 // lanes 0 and 16 hold column N=0
    int mbase = bt * 16 + half * 8;
    #pragma unroll
    for (int r = 0; r < 8; ++r)
      out[(mbase + r) * NCLS + c] = (float)acc[r];
  }
}

// ---------------------------------------------------------------------------
extern "C" void kernel_launch(void* const* d_in, const int* in_sizes, int n_in,
                              void* d_out, int out_size, void* d_ws, size_t ws_size,
                              hipStream_t stream) {
  const float* x  = (const float*)d_in[0];
  const int*   cl = (const int*)d_in[1];
  const int*   cr = (const int*)d_in[2];
  const int*   o0 = (const int*)d_in[3];
  const int*   o1 = (const int*)d_in[4];
  const int*   o2 = (const int*)d_in[5];
  const int*   o3 = (const int*)d_in[6];
  const int*   a1 = (const int*)d_in[7];
  const int*   b1 = (const int*)d_in[8];
  const int*   p1 = (const int*)d_in[9];
  const int*   a2 = (const int*)d_in[10];
  const int*   b2 = (const int*)d_in[11];
  const int*   p2 = (const int*)d_in[12];
  float* out = (float*)d_out;

  // scratch layout (all 16B-aligned offsets), total ~5.97 MB
  uint8_t*  ws    = (uint8_t*)d_ws;
  uint32_t* xpk   = (uint32_t*)(ws + 0);        //  3072*4*4  =   49152 B
  uint32_t* zconv = (uint32_t*)(ws + 49152);    // 50176*4*4  =  802816 B
  uint32_t* z1    = (uint32_t*)(ws + 851968);   // 64000*4*4  = 1024000 B
  uint8_t*  z2b   =             ws + 1875968;   // 128*32000  = 4096000 B

  k_bitpack    <<<dim3(NPIX / 256, 4), 256, 0, stream>>>(x, xpk);
  k_conv       <<<NKP / 64,            256, 0, stream>>>(xpk, cl, cr, o0, o1, o2, o3, zconv);
  k_lin        <<<dim3(NL1 / 256, 4),  256, 0, stream>>>(zconv, a1, b1, p1, z1, NL1);
  k_lin2_bytes <<<dim3(NL2 / 256, 4),  256, 0, stream>>>(z1, a2, b2, p2, z2b);
  k_groupsum_wmma<<<NCLS * 8,           32, 0, stream>>>(z2b, out);
}
